// DeformableTransformerDecoderLayer_80616536145969
// MI455X (gfx1250) — matrix operand order repaired and checked by reading, not compiled, not run on hardware
//
#include <hip/hip_runtime.h>
#include <math.h>

// ---------------------------------------------------------------------------
// MI455X (gfx1250, wave32) Deformable-DETR decoder layer.
// All GEMMs: v_wmma_f32_16x16x32_bf16, operands pre-packed to bf16 in memory
// so the K-loops are pure global_load_b128 + v_wmma. All loads of a K-step are
// issued before the first WMMA so the compiler can use partial loadcnt waits
// and overlap VMEM with matrix math.
// ---------------------------------------------------------------------------

typedef __attribute__((ext_vector_type(16))) __bf16 v16bf;
typedef __attribute__((ext_vector_type(8)))  __bf16 v8bf;
typedef __attribute__((ext_vector_type(8)))  float  v8f;

constexpr int kB   = 8;
constexpr int kLQ  = 1000;
constexpr int kD   = 256;
constexpr int kDFF = 1024;
constexpr int kNH  = 8;
constexpr int kNL  = 4;
constexpr int kNP  = 4;
constexpr int kS   = 21760;          // 128^2 + 64^2 + 32^2 + 16^2
constexpr int kR   = kB * kLQ;       // 8000 query rows
constexpr int kRS  = kB * kS;        // 174080 source rows

// ---- scalar bf16 <-> f32 helpers (epilogues / pack kernels only) ----
__device__ __forceinline__ unsigned short f2bfu(float f) {
  unsigned u = __builtin_bit_cast(unsigned, f);
  u += 0x7FFFu + ((u >> 16) & 1u);           // round-to-nearest-even
  return (unsigned short)(u >> 16);
}
__device__ __forceinline__ float bfu2f(unsigned short h) {
  unsigned u = ((unsigned)h) << 16;
  return __builtin_bit_cast(float, u);
}

__device__ __forceinline__ v8f wmma_bf16(v16bf a, v16bf b, v8f c) {
  // (neg_a, A, neg_b, B, c_mod, C, reuse_a, reuse_b)
  return __builtin_amdgcn_wmma_f32_16x16x32_bf16(false, a, false, b,
                                                 (short)0, c, false, false);
}

// Build a 16-element fragment from two aligned 8-element (16B) bf16 loads.
// A-frag (16x32 MxK, ISA 7.12.2): lane=half*16+m; p0 = row+k0+8*half,
//                                 p1 = row+k0+16+8*half.
// B-frag (32x16 KxN):             lane=half*16+n; p0 = row+k0+8*half, p1 = p0+16.
__device__ __forceinline__ v16bf ld_frag2(const __bf16* __restrict__ p0,
                                          const __bf16* __restrict__ p1) {
  v8bf lo = *reinterpret_cast<const v8bf*>(p0);
  v8bf hi = *reinterpret_cast<const v8bf*>(p1);
  return __builtin_shufflevector(lo, hi, 0, 1, 2, 3, 4, 5, 6, 7,
                                 8, 9, 10, 11, 12, 13, 14, 15);
}

// ---------------------------------------------------------------------------
// Pack f32 -> bf16 (8 elements per thread; n % 8 == 0 for all users)
// ---------------------------------------------------------------------------
__global__ __launch_bounds__(256) void f32_to_bf16_kernel(
    const float* __restrict__ in, unsigned short* __restrict__ out, int n)
{
  const int i = (blockIdx.x * blockDim.x + threadIdx.x) * 8;
  if (i >= n) return;
  const float4 a = ((const float4*)(in + i))[0];
  const float4 b = ((const float4*)(in + i))[1];
  uint4 o;
  o.x = (unsigned)f2bfu(a.x) | ((unsigned)f2bfu(a.y) << 16);
  o.y = (unsigned)f2bfu(a.z) | ((unsigned)f2bfu(a.w) << 16);
  o.z = (unsigned)f2bfu(b.x) | ((unsigned)f2bfu(b.y) << 16);
  o.w = (unsigned)f2bfu(b.z) | ((unsigned)f2bfu(b.w) << 16);
  ((uint4*)(out + i))[0] = o;
}

// ---------------------------------------------------------------------------
// o = bf16(a + b): fused residual-add + pack (feeds GEMM A operands)
// ---------------------------------------------------------------------------
__global__ __launch_bounds__(256) void add_bf16_kernel(
    const float* __restrict__ a, const float* __restrict__ b,
    unsigned short* __restrict__ o, int n)
{
  const int i = (blockIdx.x * blockDim.x + threadIdx.x) * 8;
  if (i >= n) return;
  const float4 a0 = ((const float4*)(a + i))[0];
  const float4 a1 = ((const float4*)(a + i))[1];
  const float4 b0 = ((const float4*)(b + i))[0];
  const float4 b1 = ((const float4*)(b + i))[1];
  uint4 p;
  p.x = (unsigned)f2bfu(a0.x + b0.x) | ((unsigned)f2bfu(a0.y + b0.y) << 16);
  p.y = (unsigned)f2bfu(a0.z + b0.z) | ((unsigned)f2bfu(a0.w + b0.w) << 16);
  p.z = (unsigned)f2bfu(a1.x + b1.x) | ((unsigned)f2bfu(a1.y + b1.y) << 16);
  p.w = (unsigned)f2bfu(a1.z + b1.z) | ((unsigned)f2bfu(a1.w + b1.w) << 16);
  ((uint4*)(o + i))[0] = p;
}

// ---------------------------------------------------------------------------
// GEMM: C[M,N] = act(A[M,K] @ W[N,K]^T + bias[N]); A,W bf16; out f32 or bf16.
// Block = 128 threads = 4 waves; each wave owns a 16x64 tile. All 10 b128
// loads of a K-step are issued before the first of its 4 WMMAs.
// Requires M%64==0, N%64==0, K%32==0 (true for every GEMM here).
// ---------------------------------------------------------------------------
__global__ __launch_bounds__(128) void gemm_wmma_kernel(
    const __bf16* __restrict__ A, const __bf16* __restrict__ W,
    const float* __restrict__ bias, const unsigned char* __restrict__ mask,
    float* __restrict__ Cf, unsigned short* __restrict__ Cb,
    int M, int N, int K, int relu)
{
  const int lane   = threadIdx.x & 31;
  const int wave   = threadIdx.x >> 5;
  const int lane16 = lane & 15;
  const int half   = lane >> 4;
  const int m0 = (blockIdx.y * 4 + wave) * 16;
  const int n0 = blockIdx.x * 64;

  const v8f zero = {};
  v8f acc[4];
#pragma unroll
  for (int t = 0; t < 4; ++t) acc[t] = zero;

  const __bf16* arow  = A + (size_t)(m0 + lane16) * K;
  const __bf16* brow0 = W + (size_t)(n0 + lane16) * K;

  for (int k0 = 0; k0 < K; k0 += 32) {
    __builtin_prefetch(arow + k0 + 128, 0, 1);        // global_prefetch_b8
    // issue ALL loads for this K-step first...
    const v16bf af = ld_frag2(arow + k0 + 8 * half, arow + k0 + 16 + 8 * half);
    v16bf bfr[4];
#pragma unroll
    for (int t = 0; t < 4; ++t) {
      const __bf16* br = brow0 + (size_t)t * 16 * K + k0 + 8 * half;
      bfr[t] = ld_frag2(br, br + 16);
    }
    // ...then consume with WMMAs (partial loadcnt waits, VMEM/WMMA overlap)
#pragma unroll
    for (int t = 0; t < 4; ++t) acc[t] = wmma_bf16(af, bfr[t], acc[t]);
  }

#pragma unroll
  for (int t = 0; t < 4; ++t) {
    const int n = n0 + t * 16 + lane16;
    const float bv = bias ? bias[n] : 0.f;
#pragma unroll
    for (int r = 0; r < 8; ++r) {
      const int m = m0 + r + 8 * half;                // D row = r + 8*half
      float v = acc[t][r] + bv;
      if (relu) v = fmaxf(v, 0.f);
      if (mask && mask[m]) v = 0.f;
      if (Cf) Cf[(size_t)m * N + n] = v;
      if (Cb) Cb[(size_t)m * N + n] = f2bfu(v);
    }
  }
}

// ---------------------------------------------------------------------------
// Flash-style self-attention. One wave per (b, h, query-tile of 16).
// qk: bf16 [R,512] (cols 0..255 Q proj, 256..511 K proj; 32 per head)
// v : bf16 [R,256];  o: bf16 [R,256]
// Scores/probabilities live in a 32 KB bf16 LDS strip (in-place softmax).
// ---------------------------------------------------------------------------
__global__ __launch_bounds__(32) void sa_attn_kernel(
    const __bf16* __restrict__ qk, const unsigned short* __restrict__ v,
    unsigned short* __restrict__ o)
{
  __shared__ unsigned short sc[16][1024];             // 32 KB bf16 strip

  const int wg   = blockIdx.x;                        // ((b*NH)+h)*63 + qt
  const int qt   = wg % 63;
  const int bh   = wg / 63;
  const int h    = bh % kNH;
  const int b    = bh / kNH;
  const int lane = threadIdx.x;
  const int lane16 = lane & 15;
  const int half   = lane >> 4;
  const float scale = 0.17677669529663687f;           // 1/sqrt(32)

  // ---- scores: S[16,1008] = Q_tile(16x32) x K^T, one WMMA per key tile ----
  int qrow = qt * 16 + lane16; if (qrow > kLQ - 1) qrow = kLQ - 1;
  const __bf16* qbase = qk + (size_t)(b * kLQ + qrow) * 512 + h * 32;
  const v16bf afrag = ld_frag2(qbase + 8 * half, qbase + 16 + 8 * half);
  const v8f zero = {};

  for (int kt = 0; kt < 63; ++kt) {
    int key = kt * 16 + lane16; if (key > kLQ - 1) key = kLQ - 1;
    const __bf16* kbase = qk + (size_t)(b * kLQ + key) * 512 + 256 + h * 32
                          + 8 * half;
    const v16bf bfrag = ld_frag2(kbase, kbase + 16);
    v8f c = wmma_bf16(afrag, bfrag, zero);
#pragma unroll
    for (int r = 0; r < 8; ++r)
      sc[r + 8 * half][kt * 16 + lane16] = f2bfu(c[r] * scale);
  }
  __syncthreads();

  // ---- row softmax over 1000 keys; write normalized P back as bf16 --------
  {
    const int row = lane16;
    float mx = -1e30f;
    for (int col = half; col < kLQ; col += 2) mx = fmaxf(mx, bfu2f(sc[row][col]));
    mx = fmaxf(mx, __shfl_xor(mx, 16, 32));
    float s = 0.f;
    for (int col = half; col < kLQ; col += 2) s += __expf(bfu2f(sc[row][col]) - mx);
    s += __shfl_xor(s, 16, 32);
    const float inv = 1.f / s;
    for (int col = half; col < 1024; col += 2) {
      const float p = (col < kLQ) ? __expf(bfu2f(sc[row][col]) - mx) * inv : 0.f;
      sc[row][col] = f2bfu(p);
    }
  }
  __syncthreads();

  // ---- O[16,32] = P(16x1024) x V(1024x32): 32 K-chunks x 2 N-tiles --------
  v8f acc0 = zero, acc1 = zero;
  for (int kc = 0; kc < 32; ++kc) {
    // issue the LDS P-fragment load and BOTH V fragments before the WMMAs
    const __bf16* prow = (const __bf16*)&sc[lane16][kc * 32];
    const v16bf pf = ld_frag2(prow + 16 * half, prow + 16 * half + 8); // ds_load_b128
    v16bf bfr[2];
#pragma unroll
    for (int nt = 0; nt < 2; ++nt) {
      const int n = nt * 16 + lane16;                 // dh channel
#pragma unroll
      for (int e = 0; e < 16; ++e) {
        int key = kc * 32 + 16 * half + e; if (key > kLQ - 1) key = kLQ - 1;
        bfr[nt][e] = __builtin_bit_cast(__bf16,
                         v[(size_t)(b * kLQ + key) * kD + h * 32 + n]);
      }
    }
    acc0 = wmma_bf16(pf, bfr[0], acc0);
    acc1 = wmma_bf16(pf, bfr[1], acc1);
  }

#pragma unroll
  for (int r = 0; r < 8; ++r) {
    const int qi = qt * 16 + r + 8 * half;
    if (qi < kLQ) {
      unsigned short* ob = o + (size_t)(b * kLQ + qi) * kD + h * 32;
      ob[lane16]      = f2bfu(acc0[r]);
      ob[16 + lane16] = f2bfu(acc1[r]);
    }
  }
}

// ---------------------------------------------------------------------------
// out = LayerNorm(x + res) over D=256; optional bf16 mirror for GEMM feeds.
// One wave per row, 8 elems per lane.
// ---------------------------------------------------------------------------
__global__ __launch_bounds__(256) void ln_res_kernel(
    const float* __restrict__ x, const float* __restrict__ res,
    const float* __restrict__ g, const float* __restrict__ beta,
    float* __restrict__ outf, unsigned short* __restrict__ outb, int nrows)
{
  const int row  = blockIdx.x * 8 + (threadIdx.x >> 5);
  const int lane = threadIdx.x & 31;
  if (row >= nrows) return;
  const size_t base = (size_t)row * kD + lane * 8;

  float v[8];
  float s = 0.f;
#pragma unroll
  for (int i = 0; i < 8; ++i) { v[i] = x[base + i] + res[base + i]; s += v[i]; }
#pragma unroll
  for (int o = 16; o; o >>= 1) s += __shfl_xor(s, o, 32);
  const float mean = s * (1.f / 256.f);

  float vs = 0.f;
#pragma unroll
  for (int i = 0; i < 8; ++i) { float d = v[i] - mean; vs += d * d; }
#pragma unroll
  for (int o = 16; o; o >>= 1) vs += __shfl_xor(vs, o, 32);
  const float inv = rsqrtf(vs * (1.f / 256.f) + 1e-5f);

#pragma unroll
  for (int i = 0; i < 8; ++i) {
    const float y = (v[i] - mean) * inv * g[lane * 8 + i] + beta[lane * 8 + i];
    outf[base + i] = y;
    if (outb) outb[base + i] = f2bfu(y);
  }
}

// ---------------------------------------------------------------------------
// MS-deformable sampling. One wave per (b,q,h); lane = dh channel (32).
// value is bf16; fuses the 32-way attention-weight softmax via shuffles.
// ---------------------------------------------------------------------------
__global__ __launch_bounds__(256) void deform_attn_kernel(
    const unsigned short* __restrict__ value, const float* __restrict__ offs,
    const float* __restrict__ awl, const float* __restrict__ refp,
    unsigned short* __restrict__ out)
{
  const int wid  = blockIdx.x * 8 + (threadIdx.x >> 5);
  const int lane = threadIdx.x & 31;
  const int h  = wid % kNH;
  const int bq = wid / kNH;
  const int q  = bq % kLQ;
  const int b  = bq / kLQ;
  const size_t row = (size_t)b * kLQ + q;

  // softmax over the 32 (level, point) logits for this (b,q,h)
  float wlog = awl[row * 128 + h * 32 + lane];
  float mx = wlog;
#pragma unroll
  for (int o = 16; o; o >>= 1) mx = fmaxf(mx, __shfl_xor(mx, o, 32));
  float ex = __expf(wlog - mx);
  float sm = ex;
#pragma unroll
  for (int o = 16; o; o >>= 1) sm += __shfl_xor(sm, o, 32);
  const float aw = ex / sm;

  const int Hs[4] = {128, 64, 32, 16};
  const int Ws[4] = {128, 64, 32, 16};
  const int St[4] = {0, 16384, 20480, 21504};

  float acc = 0.f;
#pragma unroll
  for (int l = 0; l < kNL; ++l) {
    const float fw = (float)Ws[l], fh = (float)Hs[l];
    const float rx = refp[(row * kNL + l) * 2 + 0];
    const float ry = refp[(row * kNL + l) * 2 + 1];
#pragma unroll
    for (int p = 0; p < kNP; ++p) {
      const float aj = __shfl(aw, l * kNP + p, 32);
      const float ox = offs[row * 256 + ((h * kNL + l) * kNP + p) * 2 + 0];
      const float oy = offs[row * 256 + ((h * kNL + l) * kNP + p) * 2 + 1];
      // loc = ref + off/(W,H); x = loc_x*W - 0.5 = rx*W + ox - 0.5
      const float x = rx * fw + ox - 0.5f;
      const float y = ry * fh + oy - 0.5f;
      const float x0 = floorf(x), y0 = floorf(y);
#pragma unroll
      for (int t = 0; t < 4; ++t) {
        const float xi = x0 + (float)(t & 1);
        const float yi = y0 + (float)(t >> 1);
        const float wt = (1.f - fabsf(x - xi)) * (1.f - fabsf(y - yi));
        const bool valid = (xi >= 0.f) && (xi < fw) && (yi >= 0.f) && (yi < fh);
        const int xc = (int)fminf(fmaxf(xi, 0.f), fw - 1.f);
        const int yc = (int)fminf(fmaxf(yi, 0.f), fh - 1.f);
        const size_t vidx =
            ((size_t)b * kS + St[l] + yc * Ws[l] + xc) * kD + h * 32 + lane;
        acc += aj * (valid ? wt : 0.f) * bfu2f(value[vidx]);
      }
    }
  }
  out[row * kD + h * 32 + lane] = f2bfu(acc);
}

// ---------------------------------------------------------------------------
// Orchestration
// ---------------------------------------------------------------------------
static inline void launch_pack(const float* in, unsigned short* out, int n,
                               hipStream_t s) {
  f32_to_bf16_kernel<<<dim3((n / 8 + 255) / 256), 256, 0, s>>>(in, out, n);
}

static inline void launch_gemm(const unsigned short* A, const unsigned short* W,
                               const float* bias, const unsigned char* mask,
                               float* Cf, unsigned short* Cb,
                               int M, int N, int K, int relu, hipStream_t s)
{
  dim3 grid(N / 64, M / 64);
  gemm_wmma_kernel<<<grid, 128, 0, s>>>((const __bf16*)A, (const __bf16*)W,
                                        bias, mask, Cf, Cb, M, N, K, relu);
}

extern "C" void kernel_launch(void* const* d_in, const int* in_sizes, int n_in,
                              void* d_out, int out_size, void* d_ws, size_t ws_size,
                              hipStream_t stream)
{
  const float* tgt      = (const float*)d_in[0];
  const float* qpos     = (const float*)d_in[1];
  const float* refp     = (const float*)d_in[2];
  const float* src      = (const float*)d_in[3];
  // d_in[4] spatial shapes, d_in[5] level starts: static, hardcoded
  const unsigned char* pad = (const unsigned char*)d_in[6];
  const float* sa_in_w  = (const float*)d_in[7];
  const float* sa_in_b  = (const float*)d_in[8];
  const float* sa_out_w = (const float*)d_in[9];
  const float* sa_out_b = (const float*)d_in[10];
  const float* ln_sa_g  = (const float*)d_in[11];
  const float* ln_sa_b  = (const float*)d_in[12];
  const float* so_w     = (const float*)d_in[13];
  const float* so_b     = (const float*)d_in[14];
  const float* aw_w     = (const float*)d_in[15];
  const float* aw_b     = (const float*)d_in[16];
  const float* vp_w     = (const float*)d_in[17];
  const float* vp_b     = (const float*)d_in[18];
  const float* op_w     = (const float*)d_in[19];
  const float* op_b     = (const float*)d_in[20];
  const float* ln_ca_g  = (const float*)d_in[21];
  const float* ln_ca_b  = (const float*)d_in[22];
  const float* w1       = (const float*)d_in[23];
  const float* b1       = (const float*)d_in[24];
  const float* w2       = (const float*)d_in[25];
  const float* b2       = (const float*)d_in[26];
  const float* ln_ff_g  = (const float*)d_in[27];
  const float* ln_ff_b  = (const float*)d_in[28];

  // ---- workspace bump allocation ----
  char* wsp = (char*)d_ws;
  auto alloc = [&](size_t bytes) -> char* {
    char* p = wsp;
    wsp += (bytes + 255) & ~(size_t)255;
    return p;
  };
  typedef unsigned short u16;

  // bf16 weight mirrors
  u16* w_sain = (u16*)alloc((size_t)768 * kD * 2);
  u16* w_saout= (u16*)alloc((size_t)kD * kD * 2);
  u16* w_so   = (u16*)alloc((size_t)256 * kD * 2);
  u16* w_aw   = (u16*)alloc((size_t)128 * kD * 2);
  u16* w_vp   = (u16*)alloc((size_t)kD * kD * 2);
  u16* w_op   = (u16*)alloc((size_t)kD * kD * 2);
  u16* w_w1   = (u16*)alloc((size_t)kDFF * kD * 2);
  u16* w_w2   = (u16*)alloc((size_t)kD * kDFF * 2);
  // bf16 activations
  u16* tgtbf  = (u16*)alloc((size_t)kR * kD * 2);
  u16* srcbf  = (u16*)alloc((size_t)kRS * kD * 2);       // 85 MB
  u16* qbuf   = (u16*)alloc((size_t)kR * kD * 2);
  u16* qkbf   = (u16*)alloc((size_t)kR * 512 * 2);
  u16* vbf    = (u16*)alloc((size_t)kR * kD * 2);
  u16* aobf   = (u16*)alloc((size_t)kR * kD * 2);
  u16* valbf  = (u16*)alloc((size_t)kRS * kD * 2);       // 85 MB
  u16* sampbf = (u16*)alloc((size_t)kR * kD * 2);
  u16* t2bf   = (u16*)alloc((size_t)kR * kD * 2);
  u16* ffn1bf = (u16*)alloc((size_t)kR * kDFF * 2);
  // f32 intermediates
  float* projf = (float*)alloc((size_t)kR * kD * 4);     // sa/ca proj, ffn2
  float* tgt1  = (float*)alloc((size_t)kR * kD * 4);
  float* tgt2  = (float*)alloc((size_t)kR * kD * 4);
  float* offsf = (float*)alloc((size_t)kR * 256 * 4);
  float* awlf  = (float*)alloc((size_t)kR * 128 * 4);

  const int nRD = kR * kD;
  const dim3 addg((nRD / 8 + 255) / 256);

  // ---- pack weights + static activations to bf16 ----
  launch_pack(sa_in_w,  w_sain,  768 * kD, stream);
  launch_pack(sa_out_w, w_saout, kD * kD, stream);
  launch_pack(so_w,     w_so,    256 * kD, stream);
  launch_pack(aw_w,     w_aw,    128 * kD, stream);
  launch_pack(vp_w,     w_vp,    kD * kD, stream);
  launch_pack(op_w,     w_op,    kD * kD, stream);
  launch_pack(w1,       w_w1,    kDFF * kD, stream);
  launch_pack(w2,       w_w2,    kD * kDFF, stream);
  launch_pack(tgt,      tgtbf,   nRD, stream);
  launch_pack(src,      srcbf,   kRS * kD, stream);

  // 1) q = bf16(tgt + query_pos)
  add_bf16_kernel<<<addg, 256, 0, stream>>>(tgt, qpos, qbuf, nRD);
  // 2) QK projection (wq;wk), 3) V projection (wv, input = tgt)
  launch_gemm(qbuf, w_sain, sa_in_b, nullptr, nullptr, qkbf, kR, 512, kD, 0, stream);
  launch_gemm(tgtbf, w_sain + (size_t)512 * kD, sa_in_b + 512, nullptr,
              nullptr, vbf, kR, kD, kD, 0, stream);
  // 4) self-attention
  sa_attn_kernel<<<dim3(kB * kNH * 63), 32, 0, stream>>>((const __bf16*)qkbf,
                                                         vbf, aobf);
  // 5) output projection, 6) tgt1 = LN(tgt + sa)
  launch_gemm(aobf, w_saout, sa_out_b, nullptr, projf, nullptr, kR, kD, kD, 0, stream);
  ln_res_kernel<<<dim3(kR / 8), 256, 0, stream>>>(projf, tgt, ln_sa_g, ln_sa_b,
                                                  tgt1, nullptr, kR);

  // 7) value projection over src (padding mask), bf16 out
  launch_gemm(srcbf, w_vp, vp_b, pad, nullptr, valbf, kRS, kD, kD, 0, stream);
  // 8) q2 = bf16(tgt1 + query_pos)
  add_bf16_kernel<<<addg, 256, 0, stream>>>(tgt1, qpos, qbuf, nRD);
  // 9) sampling offsets, 10) attention-weight logits (f32 out for sampler)
  launch_gemm(qbuf, w_so, so_b, nullptr, offsf, nullptr, kR, 256, kD, 0, stream);
  launch_gemm(qbuf, w_aw, aw_b, nullptr, awlf, nullptr, kR, 128, kD, 0, stream);
  // 11) bilinear sampling + weighted sum
  deform_attn_kernel<<<dim3(kR * kNH / 8), 256, 0, stream>>>(valbf, offsf, awlf,
                                                             refp, sampbf);
  // 12) cross-attn output projection, 13) tgt2 = LN(tgt1 + ca)
  launch_gemm(sampbf, w_op, op_b, nullptr, projf, nullptr, kR, kD, kD, 0, stream);
  ln_res_kernel<<<dim3(kR / 8), 256, 0, stream>>>(projf, tgt1, ln_ca_g, ln_ca_b,
                                                  tgt2, t2bf, kR);

  // 14) FFN
  launch_gemm(t2bf, w_w1, b1, nullptr, nullptr, ffn1bf, kR, kDFF, kD, 1, stream);
  launch_gemm(ffn1bf, w_w2, b2, nullptr, projf, nullptr, kR, kD, kDFF, 0, stream);
  // 15) out = LN(tgt2 + ffn)
  ln_res_kernel<<<dim3(kR / 8), 256, 0, stream>>>(projf, tgt2, ln_ff_g, ln_ff_b,
                                                  (float*)d_out, nullptr, kR);
}
